// LiquidNN_13812614824505
// MI455X (gfx1250) — compile-verified, matrix operand order checked by reading, not run on hardware
//
#include <hip/hip_runtime.h>

// ---------------------------------------------------------------------------
// Fused LTC kernel for MI455X (gfx1250, wave32, WMMA bf16 16x16x32).
//   grid = 8 blocks, 256 threads (8 waves). Each block owns 64 batch rows and
//   runs the full S=512 x UNFOLDS=6 recurrence with h resident in regs/LDS.
//   - All fragments served from LDS via ds_load_b128 (one-time bf16 staging).
//   - First WMMA of every accumulator chain consumes xin/bias as C operand
//     directly (separate D) -> no accumulator copy movs on the critical path.
//   - Ping-pong h buffers in LDS -> one barrier per unfold.
//   - x[:,t+1] is prefetched into registers during timestep t's unfolds, so
//     global-load latency never sits on the sequential recurrence path.
// ---------------------------------------------------------------------------

typedef __attribute__((ext_vector_type(16))) __bf16 v16bf;
typedef __attribute__((ext_vector_type(8)))  float  v8f;

struct U32x8 { unsigned int v[8]; };

__device__ __forceinline__ unsigned short f2bf(float f) {
    unsigned int u = __builtin_bit_cast(unsigned int, f);
    u += 0x7FFFu + ((u >> 16) & 1u);          // round-to-nearest-even
    return (unsigned short)(u >> 16);
}

// Hardware-transcendental tanh: 1 - 2/(e^{2x}+1).  Exact at the +-1 limits.
__device__ __forceinline__ float fast_tanh(float x) {
    float e = __expf(2.0f * x);                            // v_exp_f32
    return 1.0f - 2.0f * __builtin_amdgcn_rcpf(e + 1.0f);  // v_rcp_f32
}

// Load one 16x32 bf16 A/B fragment (this lane's 32 bytes) from LDS.
// p must point at  base + row*128 + kc*32 + hi*8  (16B aligned).
__device__ __forceinline__ v16bf frag_lds(const unsigned short* p) {
    uint4 lo  = *(const uint4*)(p);        // K +0..7   (bf16 pairs)
    uint4 hiq = *(const uint4*)(p + 16);   // K +16..23
    U32x8 u;
    u.v[0] = lo.x;  u.v[1] = lo.y;  u.v[2] = lo.z;  u.v[3] = lo.w;
    u.v[4] = hiq.x; u.v[5] = hiq.y; u.v[6] = hiq.z; u.v[7] = hiq.w;
    return __builtin_bit_cast(v16bf, u);
}

#define WMMA_BF16(A, Bm, C) \
    __builtin_amdgcn_wmma_f32_16x16x32_bf16(false, (A), false, (Bm), (short)0, (C), false, false)

__global__ __launch_bounds__(256) void ltc_fused(
    const float* __restrict__ x,    // [B,S,D]
    const float* __restrict__ Win,  // [H,D]
    const float* __restrict__ bin,  // [H]
    const float* __restrict__ Wr,   // [H,H]
    const float* __restrict__ br,   // [H]
    const float* __restrict__ Wfc,  // [1,H]
    const float* __restrict__ bfc,  // [1]
    float* __restrict__ out)        // [B]
{
    constexpr int S = 512, D = 128, H = 128, BT = 64, UNF = 6;
    constexpr float DT = 0.1f;

    // Dynamic LDS carve (112 KB total, bf16):
    //   sWin [128][128]  W_in            (32 KB)
    //   sWr  [128][128]  W_r             (32 KB)
    //   sX   [ 64][128]  x_t tile        (16 KB)
    //   sH0  [ 64][128]  h ping buffer   (16 KB)
    //   sH1  [ 64][128]  h pong buffer   (16 KB)
    extern __shared__ __align__(16) unsigned short smem[];
    unsigned short* sWin = smem;            // 16384 shorts
    unsigned short* sWr  = smem + 16384;    // 16384 shorts
    unsigned short* sX   = smem + 32768;    //  8192 shorts
    unsigned short* sH0  = smem + 40960;    //  8192 shorts
    unsigned short* sH1  = smem + 49152;    //  8192 shorts

    const int tid  = threadIdx.x;
    const int lane = tid & 31;
    const int wave = tid >> 5;
    const int ln   = lane & 15;     // lane within half-wave
    const int hi   = lane >> 4;     // half-wave selector (K-offset +8 per ISA layout)
    const int mo   = (wave >> 1) * 16;  // M-tile base (batch rows)
    const int nh   = wave & 1;          // N-half (columns 0..63 / 64..127)
    const int b0   = blockIdx.x * BT;

    // One-time staging: weights -> bf16 LDS, zero the ping h buffer.
    for (int i = tid; i < H * D; i += 256) sWin[i] = f2bf(Win[i]);
    for (int i = tid; i < H * H; i += 256) sWr[i]  = f2bf(Wr[i]);
    for (int i = tid; i < BT * H; i += 256) sH0[i] = 0;

    // Per-lane column indices + loop-invariant bias splat vectors (b_in+b_r).
    int ncol[4];
    v8f biasv[4];
    #pragma unroll
    for (int j = 0; j < 4; ++j) {
        ncol[j] = (nh * 4 + j) * 16 + ln;
        float bj = bin[ncol[j]] + br[ncol[j]];
        #pragma unroll
        for (int r = 0; r < 8; ++r) biasv[j][r] = bj;
    }

    float hreg[4][8];                       // h in C-layout fp32
    #pragma unroll
    for (int j = 0; j < 4; ++j)
        #pragma unroll
        for (int r = 0; r < 8; ++r) hreg[j][r] = 0.0f;

    const int arow = mo + ln;               // this lane's A-row (within tile)
    const int koff = hi * 8;                // per-half K offset
    const int srow = (mo + hi * 8) * H;     // C-layout store row base

    // x prefetch pipeline: this thread's 8 float4s of the x_t tile.
    // q = tid + ii*256 -> row = q>>5 (0..63), kq = q&31 (float4 within row).
    // One wave covers exactly one 128-float row -> fully coalesced.
    float4 xf[8];
    #pragma unroll
    for (int ii = 0; ii < 8; ++ii) {
        int q = tid + ii * 256, row = q >> 5, kq = q & 31;
        xf[ii] = *(const float4*)(x + ((size_t)(b0 + row) * S + 0) * D + kq * 4);
    }

    __syncthreads();

    #pragma unroll 1
    for (int t = 0; t < S; ++t) {
        // ---- commit prefetched x_t tile -> bf16 LDS (b64 packed stores) ---
        #pragma unroll
        for (int ii = 0; ii < 8; ++ii) {
            int q = tid + ii * 256, row = q >> 5, kq = q & 31;
            float4 v = xf[ii];
            unsigned d0 = (unsigned)f2bf(v.x) | ((unsigned)f2bf(v.y) << 16);
            unsigned d1 = (unsigned)f2bf(v.z) | ((unsigned)f2bf(v.w) << 16);
            *(uint2*)(sX + row * D + kq * 4) = make_uint2(d0, d1);
        }
        __syncthreads();

        // ---- issue next timestep's x loads; latency hides under unfolds ---
        {
            int tn = (t + 1 < S) ? (t + 1) : t;   // clamped (last iter harmless)
            #pragma unroll
            for (int ii = 0; ii < 8; ++ii) {
                int q = tid + ii * 256, row = q >> 5, kq = q & 31;
                xf[ii] = *(const float4*)(x + ((size_t)(b0 + row) * S + tn) * D + kq * 4);
            }
        }

        // ---- xin = x_t @ W_in^T + (b_in + b_r)  (16 WMMAs) ----------------
        v8f xin[4];
        {
            v16bf a[4];
            #pragma unroll
            for (int kc = 0; kc < 4; ++kc)
                a[kc] = frag_lds(sX + arow * D + kc * 32 + koff);
            #pragma unroll
            for (int j = 0; j < 4; ++j) {
                v16bf w = frag_lds(sWin + ncol[j] * D + koff);
                xin[j] = WMMA_BF16(a[0], w, biasv[j]);
            }
            #pragma unroll
            for (int kc = 1; kc < 4; ++kc)
                #pragma unroll
                for (int j = 0; j < 4; ++j) {
                    v16bf w = frag_lds(sWin + ncol[j] * D + kc * 32 + koff);
                    xin[j] = WMMA_BF16(a[kc], w, xin[j]);
                }
        }

        // ---- 6 ODE unfolds: v=tanh(xin + h@Wr^T); h = 0.9h + 0.1v ---------
        // Ping-pong h buffers: even unfolds read sH0/write sH1, odd reverse.
        // UNF is even, so every timestep begins reading sH0.
        #pragma unroll 2
        for (int u = 0; u < UNF; ++u) {
            const unsigned short* sHr = (u & 1) ? sH1 : sH0;
            unsigned short*       sHw = (u & 1) ? sH0 : sH1;

            v16bf a[4];
            #pragma unroll
            for (int kc = 0; kc < 4; ++kc)
                a[kc] = frag_lds(sHr + arow * H + kc * 32 + koff);

            v8f c[4];
            #pragma unroll
            for (int j = 0; j < 4; ++j) {
                v16bf w = frag_lds(sWr + ncol[j] * H + koff);
                c[j] = WMMA_BF16(a[0], w, xin[j]);   // seed from xin, no copy
            }
            #pragma unroll
            for (int kc = 1; kc < 4; ++kc)
                #pragma unroll
                for (int j = 0; j < 4; ++j) {
                    v16bf w = frag_lds(sWr + ncol[j] * H + kc * 32 + koff);
                    c[j] = WMMA_BF16(a[kc], w, c[j]);
                }

            #pragma unroll
            for (int j = 0; j < 4; ++j)
                #pragma unroll
                for (int r = 0; r < 8; ++r) {
                    hreg[j][r] = (1.0f - DT) * hreg[j][r] + DT * fast_tanh(c[j][r]);
                    sHw[srow + r * H + ncol[j]] = f2bf(hreg[j][r]);
                }
            __syncthreads();   // new h buffer visible; old buffer free to reuse
        }
    }

    // ---- readout: out[b] = h[b,:] . W_fc[0,:] + b_fc  (fp32 h via LDS) ----
    float* sF = (float*)smem;  // weights no longer needed; reuse as fp32 h
    #pragma unroll
    for (int j = 0; j < 4; ++j)
        #pragma unroll
        for (int r = 0; r < 8; ++r)
            sF[(mo + r + hi * 8) * H + ncol[j]] = hreg[j][r];
    __syncthreads();

    if (tid < BT) {
        float acc = bfc[0];
        #pragma unroll 4
        for (int k = 0; k < H; ++k) acc += sF[tid * H + k] * Wfc[k];
        out[b0 + tid] = acc;
    }
}

extern "C" void kernel_launch(void* const* d_in, const int* in_sizes, int n_in,
                              void* d_out, int out_size, void* d_ws, size_t ws_size,
                              hipStream_t stream) {
    (void)in_sizes; (void)n_in; (void)d_ws; (void)ws_size; (void)out_size;
    const float* x   = (const float*)d_in[0];
    const float* Win = (const float*)d_in[1];
    const float* bin = (const float*)d_in[2];
    const float* Wr  = (const float*)d_in[3];
    const float* br  = (const float*)d_in[4];
    const float* Wfc = (const float*)d_in[5];
    const float* bfc = (const float*)d_in[6];
    float* out = (float*)d_out;

    constexpr size_t kLds = 112u * 1024u;  // sWin + sWr + sX + sH0 + sH1 (bf16)
    ltc_fused<<<dim3(8), dim3(256), kLds, stream>>>(x, Win, bin, Wr, br, Wfc, bfc, out);
}
